// FullBatchCrossEntropyLoss_78125455114493
// MI455X (gfx1250) — matrix operand order repaired and checked by reading, not compiled, not run on hardware
//
#include <hip/hip_runtime.h>

// ---- CDNA5 vector types ----
typedef __attribute__((ext_vector_type(16))) __bf16 v16bf;
typedef __attribute__((ext_vector_type(8)))  __bf16 v8bf;
typedef __attribute__((ext_vector_type(4)))  __bf16 v4bf;
typedef __attribute__((ext_vector_type(8)))  float  v8f;

#define D_DIM  128
#define S_DIM  1024
#define V_DIM  100000
#define N_LBL  4096
#define KTILES (V_DIM / 16)   // 6250, exact

#define LOG2E_F 1.4426950408889634f
#define LN2_F   0.6931471805599453f

union AFrag { v16bf v; v8bf h[2]; };

__device__ __forceinline__ float wave_reduce_sum32(float v) {
  v += __shfl_xor(v, 1, 32);
  v += __shfl_xor(v, 2, 32);
  v += __shfl_xor(v, 4, 32);
  v += __shfl_xor(v, 8, 32);
  v += __shfl_xor(v, 16, 32);
  return v;
}

// ---------- Kernel 1: L2-normalize keys, f32 -> bf16 (one wave per row) ----------
__global__ void __launch_bounds__(256)
nrm_keys(const float* __restrict__ keys, unsigned short* __restrict__ outp) {
  __bf16* out = (__bf16*)outp;
  int wid  = (blockIdx.x * blockDim.x + threadIdx.x) >> 5;
  int lane = threadIdx.x & 31;
  if (wid >= V_DIM) return;
  const float4* src = (const float4*)(keys + (size_t)wid * D_DIM);
  float4 x = src[lane];
  float ss = x.x * x.x + x.y * x.y + x.z * x.z + x.w * x.w;
  ss = wave_reduce_sum32(ss);
  float inv = 1.0f / fmaxf(sqrtf(ss), 1e-12f);
  v4bf o;
  o[0] = (__bf16)(x.x * inv);
  o[1] = (__bf16)(x.y * inv);
  o[2] = (__bf16)(x.z * inv);
  o[3] = (__bf16)(x.w * inv);
  ((v4bf*)(out + (size_t)wid * D_DIM))[lane] = o;
}

// ---------- Kernel 2: gather + L2-normalize queries -> bf16, PRE-SCALED by log2(e) ----------
// Storing q_norm * log2e makes the WMMA logits base-2, so the softmax inner loop
// needs only v_exp_f32 (no per-element multiply by log2e).
__global__ void __launch_bounds__(256)
gather_nrm_q(const float* __restrict__ q, const long long* __restrict__ locs,
             unsigned short* __restrict__ outp) {
  __bf16* out = (__bf16*)outp;
  int wid  = (blockIdx.x * blockDim.x + threadIdx.x) >> 5;
  int lane = threadIdx.x & 31;
  if (wid >= N_LBL) return;
  long long b = locs[2 * wid + 0];
  long long s = locs[2 * wid + 1];
  const float4* src = (const float4*)(q + ((size_t)b * S_DIM + (size_t)s) * D_DIM);
  float4 x = src[lane];
  float ss = x.x * x.x + x.y * x.y + x.z * x.z + x.w * x.w;
  ss = wave_reduce_sum32(ss);
  float inv = LOG2E_F / fmaxf(sqrtf(ss), 1e-12f);
  v4bf o;
  o[0] = (__bf16)(x.x * inv);
  o[1] = (__bf16)(x.y * inv);
  o[2] = (__bf16)(x.z * inv);
  o[3] = (__bf16)(x.w * inv);
  ((v4bf*)(out + (size_t)wid * D_DIM))[lane] = o;
}

// ---------- Kernel 3: WMMA logits + exp2-sum over all V keys ----------
// Block = 128 threads (4 waves). Block owns 32 query rows (two 16-row WMMA tiles).
// Each wave strides over key tiles of 16; A fragments stay in registers.
// Cosine logits are bounded (|l*log2e| <= 1.45) -> no online max, no exp range guard:
// raw v_exp_f32 via __builtin_amdgcn_exp2f.
__global__ void __launch_bounds__(128)
nce_main(const unsigned short* __restrict__ qnp,
         const unsigned short* __restrict__ knp,
         float* __restrict__ logZ) {
  const __bf16* qn = (const __bf16*)qnp;
  const __bf16* kn = (const __bf16*)knp;
  __shared__ float partial[4][32];

  int lane = threadIdx.x & 31;
  int wave = threadIdx.x >> 5;
  int half = lane >> 4;    // K-half selector (contiguous A/B k order: k = 16*half + s)
  int sub  = lane & 15;    // M (A) or N (B) within tile
  int n0   = blockIdx.x * 32;

  // Preload A fragments: 2 row-tiles x 4 K-chunks of 32.
  // Lane holds row M=sub; elems 0..15 = K(khalf*16 + 0..15).
  AFrag a[2][4];
#pragma unroll
  for (int mt = 0; mt < 2; ++mt) {
    const __bf16* qrow = qn + (size_t)(n0 + mt * 16 + sub) * D_DIM + half * 16;
#pragma unroll
    for (int kc = 0; kc < 4; ++kc) {
      a[mt][kc].h[0] = *(const v8bf*)(qrow + kc * 32);
      a[mt][kc].h[1] = *(const v8bf*)(qrow + kc * 32 + 8);
    }
  }

  float s[2][8];
#pragma unroll
  for (int mt = 0; mt < 2; ++mt)
#pragma unroll
    for (int r = 0; r < 8; ++r) s[mt][r] = 0.0f;

  for (int vt = wave; vt < KTILES; vt += 4) {
    // B fragment: lane holds key column N=sub, 16 consecutive K at khalf*16 (contiguous 32B).
    const __bf16* kb = kn + (size_t)(vt * 16 + sub) * D_DIM + half * 16;
    v16bf b[4];
#pragma unroll
    for (int kc = 0; kc < 4; ++kc) b[kc] = *(const v16bf*)(kb + kc * 32);

#pragma unroll
    for (int mt = 0; mt < 2; ++mt) {
      v8f acc = {};
#pragma unroll
      for (int kc = 0; kc < 4; ++kc)
        acc = __builtin_amdgcn_wmma_f32_16x16x32_bf16(
            false, a[mt][kc].v, false, b[kc], (short)0, acc, false, false);
      // acc[r] = log2e * logit(row = r + 8*half, key = vt*16 + sub)
#pragma unroll
      for (int r = 0; r < 8; ++r)
        s[mt][r] += __builtin_amdgcn_exp2f(acc[r]);   // bare v_exp_f32, no range guard
    }
  }

  // Reduce over the 16 key columns of each half-wave (xor masks < 16 stay in-half).
#pragma unroll
  for (int mt = 0; mt < 2; ++mt)
#pragma unroll
    for (int r = 0; r < 8; ++r) {
      float v = s[mt][r];
      v += __shfl_xor(v, 1, 32);
      v += __shfl_xor(v, 2, 32);
      v += __shfl_xor(v, 4, 32);
      v += __shfl_xor(v, 8, 32);
      s[mt][r] = v;
    }

  // Deterministic cross-wave combine via LDS (no float atomics).
  if (sub == 0) {
#pragma unroll
    for (int mt = 0; mt < 2; ++mt)
#pragma unroll
      for (int r = 0; r < 8; ++r)
        partial[wave][mt * 16 + half * 8 + r] = s[mt][r];
  }
  __syncthreads();
  if (threadIdx.x < 32) {
    int i = threadIdx.x;
    float t = partial[0][i] + partial[1][i] + partial[2][i] + partial[3][i];
    logZ[n0 + i] = logf(t);   // s = sum(e^logit) exactly; logZ = ln(s)
  }
}

// ---------- Kernel 4: true logit + per-label loss (one wave per label) ----------
// qn is pre-scaled by log2e, so dot(qn,kn) = logit*log2e -> multiply by ln2.
__global__ void __launch_bounds__(256)
true_logit_loss(const unsigned short* __restrict__ qnp,
                const unsigned short* __restrict__ knp,
                const long long* __restrict__ labels,
                const float* __restrict__ logZ,
                float* __restrict__ lossn) {
  int wid  = (blockIdx.x * blockDim.x + threadIdx.x) >> 5;
  int lane = threadIdx.x & 31;
  if (wid >= N_LBL) return;
  const __bf16* qr = (const __bf16*)qnp + (size_t)wid * D_DIM;
  const __bf16* kr = (const __bf16*)knp + (size_t)labels[wid] * D_DIM;
  v4bf qv = ((const v4bf*)qr)[lane];
  v4bf kv = ((const v4bf*)kr)[lane];
  float d = 0.0f;
#pragma unroll
  for (int r = 0; r < 4; ++r) d += (float)qv[r] * (float)kv[r];
  d = wave_reduce_sum32(d);
  if (lane == 0) lossn[wid] = logZ[wid] - d * LN2_F;
}

// ---------- Kernel 5: deterministic mean reduction ----------
__global__ void __launch_bounds__(256)
reduce_mean(const float* __restrict__ lossn, float* __restrict__ out) {
  __shared__ float red[256];
  float t = 0.0f;
  for (int i = threadIdx.x; i < N_LBL; i += 256) t += lossn[i];
  red[threadIdx.x] = t;
  __syncthreads();
  for (int s = 128; s > 0; s >>= 1) {
    if ((int)threadIdx.x < s) red[threadIdx.x] += red[threadIdx.x + s];
    __syncthreads();
  }
  if (threadIdx.x == 0) out[0] = red[0] / (float)N_LBL;
}

extern "C" void kernel_launch(void* const* d_in, const int* in_sizes, int n_in,
                              void* d_out, int out_size, void* d_ws, size_t ws_size,
                              hipStream_t stream) {
  const float*     qemb   = (const float*)d_in[0];       // (16,1024,128) f32
  const float*     kemb   = (const float*)d_in[1];       // (100000,128)  f32
  const long long* locs   = (const long long*)d_in[2];   // (4096,2) i64
  const long long* labels = (const long long*)d_in[3];   // (4096,)  i64
  float*           out    = (float*)d_out;               // scalar f32

  // Workspace layout (all offsets 256B-aligned):
  char* ws = (char*)d_ws;
  unsigned short* kn   = (unsigned short*)ws;                         // 100000*128*2 = 25,600,000 B
  unsigned short* qn   = (unsigned short*)(ws + 25600000);            //   4096*128*2 =  1,048,576 B
  float*          logZ = (float*)(ws + 25600000 + 1048576);           //   4096*4
  float*          lossn= logZ + N_LBL;                                //   4096*4

  (void)in_sizes; (void)n_in; (void)out_size; (void)ws_size;

  nrm_keys       <<<V_DIM / 8, 256, 0, stream>>>(kemb, kn);           // 12500 blocks, 8 waves ea
  gather_nrm_q   <<<N_LBL / 8, 256, 0, stream>>>(qemb, locs, qn);     // 512 blocks
  nce_main       <<<N_LBL / 32, 128, 0, stream>>>(qn, kn, logZ);      // 128 blocks x 4 waves
  true_logit_loss<<<N_LBL / 8, 256, 0, stream>>>(qn, kn, labels, logZ, lossn);
  reduce_mean    <<<1, 256, 0, stream>>>(lossn, out);
}
